// Canonical_41609643163873
// MI455X (gfx1250) — matrix operand order repaired and checked by reading, not compiled, not run on hardware
//
#include <hip/hip_runtime.h>
#include <math.h>
#include <stdint.h>

typedef __attribute__((ext_vector_type(16))) _Float16 v16h;
typedef __attribute__((ext_vector_type(8)))  _Float16 v8h;
typedef __attribute__((ext_vector_type(8)))  float    v8f;

#define NPTS   8192          // 2 * 64 * 64 downsampled points
#define NC     19            // classes
#define CP     32            // padded K for one 16x16x32 WMMA
#define NT     (NPTS / 16)   // 512 column tiles
#define DF     8
#define HH     512
#define WW     512
#define INV_BW 50.0f         // 1 / 0.02

// ---------------------------------------------------------------------------
// Phase 1: fused strided-softmax + log-softmax + Dirichlet log-normalizer +
// target argmax. One thread per downsampled point.
// ---------------------------------------------------------------------------
__global__ void __launch_bounds__(128) ece_prep(
    const float* __restrict__ inp, const float* __restrict__ tgt,
    _Float16* __restrict__ Ah,     // [NPTS][32] log f  (f16, zero-padded)
    _Float16* __restrict__ Bh,     // [NPTS][32] f/bw   (f16, zero-padded)
    float* __restrict__ Ff,        // [NPTS][19] f      (f32, for final L2)
    float* __restrict__ logb,      // [NPTS] log Beta normalizer
    int* __restrict__ yv)          // [NPTS] argmax label
{
    int n = blockIdx.x * blockDim.x + threadIdx.x;
    if (n >= NPTS) return;
    int b  = n >> 12;
    int hw = n & 4095;
    int h  = (hw >> 6) * DF;
    int w  = (hw & 63) * DF;
    const size_t cs   = (size_t)HH * WW;
    const size_t base = (size_t)b * NC * cs + (size_t)h * WW + w;

    float logits[NC];
    float m = -3.4e38f;
#pragma unroll
    for (int c = 0; c < NC; ++c) {
        logits[c] = inp[base + (size_t)c * cs];
        m = fmaxf(m, logits[c]);
    }
    float se = 0.f;
#pragma unroll
    for (int c = 0; c < NC; ++c) se += __expf(logits[c] - m);
    float lse = __logf(se);

    float lgsum = 0.f, asum = 0.f;
#pragma unroll
    for (int c = 0; c < NC; ++c) {
        float lf = logits[c] - m - lse;   // log f
        float f  = __expf(lf);            // f (softmax prob)
        Ff[n * NC + c] = f;
        Ah[n * CP + c] = (_Float16)lf;
        Bh[n * CP + c] = (_Float16)(f * INV_BW);
        float alpha = f * INV_BW + 1.0f;
        asum  += alpha;
        lgsum += lgammaf(alpha);
    }
#pragma unroll
    for (int c = NC; c < CP; ++c) {
        Ah[n * CP + c] = (_Float16)0.f;
        Bh[n * CP + c] = (_Float16)0.f;
    }
    logb[n] = lgsum - lgammaf(asum);

    float bm = tgt[base];
    int   bi = 0;
#pragma unroll
    for (int c = 1; c < NC; ++c) {
        float t = tgt[base + (size_t)c * cs];
        if (t > bm) { bm = t; bi = c; }
    }
    yv[n] = bi;
}

// ---------------------------------------------------------------------------
// Phase 2: pairwise kernel. One wave per 16-row tile, 4 waves per block.
// Triple-buffered async DMA (GLOBAL_LOAD_ASYNC_TO_LDS_B128/B32, ASYNCcnt)
// with the 3-stage pipeline unrolled x3 so every slot index is compile-time
// constant (no register rotation, immediate ds offsets). Steady-state loop:
// LDS loads -> v_wmma_f32_16x16x32_f16 -> exp -> ds_add_f32 scatter, one
// barrier per tile, no global traffic, no diagonal masking (self term
// subtracted once in the epilogue).
// ---------------------------------------------------------------------------
__global__ void __launch_bounds__(128) ece_pair(
    const _Float16* __restrict__ Ah, const _Float16* __restrict__ Bh,
    const float* __restrict__ logb,  const int* __restrict__ yv,
    const float* __restrict__ Ff,    float* __restrict__ lossp)
{
    __shared__ float ky[4][16][20];                        // per-wave [row][class]
    __shared__ __align__(32) _Float16 Bst[3][16 * CP];     // triple-buffered B tile
    __shared__ float lbst[3][16];                          // staged log_beta
    __shared__ int   yst[3][16];                           // staged labels

    const int lane   = threadIdx.x & 31;
    const int wid    = threadIdx.x >> 5;
    const int tile   = blockIdx.x * 4 + wid;
    const int base_i = tile * 16;
    const int r      = lane & 15;
    const int mrow   = (lane >> 4) * 8;   // D rows this lane holds: mrow..mrow+7

    for (int t = lane; t < 16 * 20; t += 32) ky[wid][t / 20][t % 20] = 0.f;

    // A fragment: row = lane&15; halves 0..7 = K ho..ho+7, 8..15 = K ho+8..ho+15
    const int ho = (lane >> 4) * 16;
    const v8h alo = *(const v8h*)(Ah + (size_t)(base_i + r) * CP + ho);
    const v8h ahi = *(const v8h*)(Ah + (size_t)(base_i + r) * CP + 8 + ho);
    v16h a;
#pragma unroll
    for (int t = 0; t < 8; ++t) { a[t] = alo[t]; a[t + 8] = ahi[t]; }

    // --- async staging setup: per tile = 1024B B (32 lanes x 2 x b128) + 64B
    //     log_beta (lanes 0-15) + 64B y (lanes 16-31) in one b32 DMA.
    uint32_t wB[3], wS[3];
#pragma unroll
    for (int s = 0; s < 3; ++s) {
        wB[s] = (uint32_t)(uintptr_t)(&Bst[s][0]) + (uint32_t)(lane * 32);
        wS[s] = (lane < 16) ? (uint32_t)(uintptr_t)(&lbst[s][lane])
                            : (uint32_t)(uintptr_t)(&yst[s][lane - 16]);
    }
    const uint64_t gB0 = (uint64_t)(uintptr_t)Bh + (uint64_t)(lane * 32);
    const uint64_t gS0 = (lane < 16) ? (uint64_t)(uintptr_t)(logb + lane)
                                     : (uint64_t)(uintptr_t)(yv + (lane - 16));
    if (wid == 0) {   // prologue: DMA tiles 0 and 1 into slots 0 and 1
#pragma unroll
        for (int s = 0; s < 2; ++s) {
            const uint64_t g  = gB0 + (uint64_t)s * 1024;
            const uint64_t gs = gS0 + (uint64_t)s * 64;
            asm volatile(
                "global_load_async_to_lds_b128 %0, %2, off\n\t"
                "global_load_async_to_lds_b128 %0, %2, off offset:16\n\t"
                "global_load_async_to_lds_b32  %1, %3, off"
                :: "v"(wB[s]), "v"(wS[s]), "v"(g), "v"(gs) : "memory");
        }
    }
    uint64_t gBn = gB0 + 2 * 1024;   // next B tile to DMA (tile 2)
    uint64_t gSn = gS0 + 2 * 64;

    // One pipeline stage: consume slot S (= tile jt), optionally DMA tile
    // jt+2 into slot (S+2)%3. ISSUE/LAST are compile-time peel flags.
#define STEP(S, LAST, ISSUE)                                                   \
    {                                                                          \
        if (wid == 0) {                                                        \
            if (LAST) asm volatile("s_wait_asynccnt 0x0" ::: "memory");        \
            else      asm volatile("s_wait_asynccnt 0x3" ::: "memory");        \
        }                                                                      \
        __syncthreads();                   /* slot S published to all waves */ \
        if (ISSUE) {                                                           \
            if (wid == 0) {                                                    \
                asm volatile(                                                  \
                    "global_load_async_to_lds_b128 %0, %2, off\n\t"            \
                    "global_load_async_to_lds_b128 %0, %2, off offset:16\n\t"  \
                    "global_load_async_to_lds_b32  %1, %3, off"                \
                    :: "v"(wB[((S) + 2) % 3]), "v"(wS[((S) + 2) % 3]),         \
                       "v"(gBn), "v"(gSn) : "memory");                         \
                gBn += 1024;                                                   \
                gSn += 64;                                                     \
            }                                                                  \
        }                                                                      \
        const v16h bb =                                                        \
            *(const v16h*)(&Bst[(S)][r * CP + (lane >> 4) * 16]);              \
        const float lb = lbst[(S)][r];                                         \
        const int   yj = yst[(S)][r];                                          \
        v8f acc = {};                                                          \
        acc = __builtin_amdgcn_wmma_f32_16x16x32_f16(                          \
            false, a, false, bb, (short)0, acc, false, false);                 \
        _Pragma("unroll")                                                      \
        for (int m = 0; m < 8; ++m) {                                          \
            const float k = __expf(acc[m] - lb);                               \
            atomicAdd(&ky[wid][mrow + m][yj], k);  /* ds_add_f32 */            \
        }                                                                      \
    }

    // NT-2 = 510 = 3*170: main loop fully in steady state, 2-tile epilogue.
    for (int jt = 0; jt < NT - 2; jt += 3) {
        STEP(0, false, true)
        STEP(1, false, true)
        STEP(2, false, true)
    }
    STEP(0, false, false)   // tile NT-2: tile NT-1 still in flight, no issue
    STEP(1, true,  false)   // tile NT-1: drain
#undef STEP

    if (lane < 16) {
        const int i = base_i + lane;
        // subtract the self (diagonal) kernel from its class bucket, using the
        // same f16 operands the WMMA consumed so the cancellation is tight
        float sdot = 0.f;
#pragma unroll
        for (int c = 0; c < NC; ++c)
            sdot += (float)Ah[(size_t)i * CP + c] * (float)Bh[(size_t)i * CP + c];
        const int yi = yv[i];
        ky[wid][lane][yi] -= __expf(sdot - logb[i]);

        float den = 0.f;
#pragma unroll
        for (int c = 0; c < NC; ++c) den += ky[wid][lane][c];
        den = fmaxf(den, 1e-10f);
        float loss = 0.f;
#pragma unroll
        for (int c = 0; c < NC; ++c) {
            float d = ky[wid][lane][c] / den - Ff[i * NC + c];
            loss += d * d;
        }
        lossp[i] = loss;
    }
}

// ---------------------------------------------------------------------------
// Phase 3: deterministic single-block tree reduction -> mean.
// ---------------------------------------------------------------------------
__global__ void __launch_bounds__(256) ece_reduce(
    const float* __restrict__ lossp, float* __restrict__ out)
{
    __shared__ float sm[256];
    float s = 0.f;
    for (int t = threadIdx.x; t < NPTS; t += 256) s += lossp[t];
    sm[threadIdx.x] = s;
    __syncthreads();
    for (int st = 128; st > 0; st >>= 1) {
        if ((int)threadIdx.x < st) sm[threadIdx.x] += sm[threadIdx.x + st];
        __syncthreads();
    }
    if (threadIdx.x == 0) out[0] = sm[0] / (float)NPTS;
}

// ---------------------------------------------------------------------------
extern "C" void kernel_launch(void* const* d_in, const int* in_sizes, int n_in,
                              void* d_out, int out_size, void* d_ws, size_t ws_size,
                              hipStream_t stream) {
    (void)in_sizes; (void)n_in; (void)out_size; (void)ws_size;
    const float* inp = (const float*)d_in[0];
    const float* tgt = (const float*)d_in[1];
    float* out = (float*)d_out;

    char* ws = (char*)d_ws;
    size_t off = 0;
    _Float16* Ah   = (_Float16*)(ws + off); off += (size_t)NPTS * CP * sizeof(_Float16); // 512 KB
    _Float16* Bh   = (_Float16*)(ws + off); off += (size_t)NPTS * CP * sizeof(_Float16); // 512 KB
    float*    Ff   = (float*)   (ws + off); off += (size_t)NPTS * NC * sizeof(float);    // 608 KB
    float*    logb = (float*)   (ws + off); off += (size_t)NPTS * sizeof(float);
    int*      yv   = (int*)     (ws + off); off += (size_t)NPTS * sizeof(int);
    float*    lossp= (float*)   (ws + off); off += (size_t)NPTS * sizeof(float);

    ece_prep<<<NPTS / 128, 128, 0, stream>>>(inp, tgt, Ah, Bh, Ff, logb, yv);
    ece_pair<<<NT / 4, 128, 0, stream>>>(Ah, Bh, logb, yv, Ff, lossp);
    ece_reduce<<<1, 256, 0, stream>>>(lossp, out);
}
